// UnpitchedChannelsEncoder_65755949302437
// MI455X (gfx1250) — matrix operand order repaired and checked by reading, not compiled, not run on hardware
//
#include <hip/hip_runtime.h>
#include <hip/hip_bf16.h>

typedef _Float16 f16;
typedef __attribute__((ext_vector_type(16))) _Float16 v16h;
typedef __attribute__((ext_vector_type(8)))  float    v8f;

#define NTOK   32768   // B*BARS*BEATS*FRAC tokens
#define INF    940
#define NOUT   100
#define XSTRIDE 128    // padded f16 activation stride (zeros in 100..127)
#define G2     64      // 4*BEAT_H
#define H2     16
#define G3     128     // 4*BAR_H
#define H3     32
#define BEATS_ELEMS (32768*16)   // 524288 floats (first output)

__device__ __forceinline__ float sigf(float x) { return 1.0f / (1.0f + __expf(-x)); }

// ---------------------------------------------------------------------------
// Kernel 1: fused permute([10,2,47]->[10,47,2]) + Linear(940->100) + ReLU
// WMMA f16 GEMM, M=32768 N=112(pad) K=960(pad). Stores x as f16 [NTOK][128].
// ---------------------------------------------------------------------------
__global__ __launch_bounds__(256) void k_linear_relu(
    const float* __restrict__ chan, const float* __restrict__ lin_w,
    const float* __restrict__ lin_b, f16* __restrict__ xws)
{
  __shared__ f16 sB[112 * 34];            // weight K-chunk, padded stride (17 banks)
  const int tid  = threadIdx.x;
  const int lane = tid & 31;
  const int wave = tid >> 5;
  const int t0   = (blockIdx.x * 8 + wave) * 16;   // 16-token M-tile per wave
  const int col  = lane & 15;                      // A row / B col for this lane
  const long abase = (long)(t0 + col) * INF;
  const int  akb = (lane < 16) ? 0 : 8;
  const int  bkb = (lane < 16) ? 0 : 16;

  v8f acc[7];
  #pragma unroll
  for (int n = 0; n < 7; ++n)
    #pragma unroll
    for (int r = 0; r < 8; ++r) acc[n][r] = 0.0f;

  for (int kc = 0; kc < 30; ++kc) {
    const int k0 = kc * 32;
    __syncthreads();                       // protect sB from previous iter's readers
    // cooperatively stage lin_w[:, k0:k0+32] as f16 (zero-padded)
    for (int i = tid; i < 112 * 32; i += 256) {
      const int o = i >> 5, kk = i & 31;
      const int f = k0 + kk;
      float w = (o < NOUT && f < INF) ? lin_w[o * INF + f] : 0.0f;
      sB[o * 34 + kk] = (f16)w;
    }
    __syncthreads();

    // A fragment: per-lane gather with the [10,2,47] -> [10,47,2] remap
    v16h a;
    #pragma unroll
    for (int e = 0; e < 16; ++e) {
      const int K = e + bkb;
      const int f = k0 + K;
      float v = 0.0f;
      if (f < INF) {
        const int c  = f / 94;
        const int r  = f - c * 94;
        const int j  = r & 1;     // inner channel pair index (memory stride 47)
        const int kk = r >> 1;    // 47-dim index (memory stride 1)
        v = chan[abase + c * 94 + j * 47 + kk];
      }
      a[e] = (f16)v;
    }
    if (kc < 28) __builtin_prefetch(chan + abase + (k0 + 32), 0, 0);

    // 7 N-tiles of v_wmma_f32_16x16x32_f16 (A reused across tiles)
    #pragma unroll
    for (int n = 0; n < 7; ++n) {
      v16h b;
      const f16* bp = &sB[(n * 16 + col) * 34 + bkb];
      #pragma unroll
      for (int e = 0; e < 16; ++e) b[e] = bp[e];
      acc[n] = __builtin_amdgcn_wmma_f32_16x16x32_f16(
          false, a, false, b, (short)0, acc[n], false, false);
    }
  }

  // epilogue: +bias, ReLU, convert to f16, store padded row (zeros 100..127)
  const int rbase = (lane < 16) ? 0 : 8;
  #pragma unroll
  for (int n = 0; n < 7; ++n) {
    const int o = n * 16 + col;
    #pragma unroll
    for (int r = 0; r < 8; ++r) {
      float v = 0.0f;
      if (o < NOUT) {
        v = acc[n][r] + lin_b[o];
        v = v > 0.0f ? v : 0.0f;
      }
      xws[(long)(t0 + rbase + r) * XSTRIDE + o] = (f16)v;
    }
  }
  {
    const int o = 112 + col;
    #pragma unroll
    for (int r = 0; r < 8; ++r)
      xws[(long)(t0 + rbase + r) * XSTRIDE + o] = (f16)0.0f;
  }
}

// ---------------------------------------------------------------------------
// Kernel 2: xg = x @ b_wih.T + (bih+bhh).  WMMA, M=32768 N=64 K=128(pad).
// ---------------------------------------------------------------------------
__global__ __launch_bounds__(256) void k_beats_xg(
    const f16* __restrict__ xws, const float* __restrict__ b_wih,
    const float* __restrict__ b_bih, const float* __restrict__ b_bhh,
    float* __restrict__ xg)
{
  __shared__ f16 sW[64 * 130];   // full b_wih as f16, K padded 100->128, stride 130
  const int tid = threadIdx.x;
  for (int i = tid; i < 64 * 128; i += 256) {
    const int o = i >> 7, k = i & 127;
    float w = (k < NOUT) ? b_wih[o * NOUT + k] : 0.0f;
    sW[o * 130 + k] = (f16)w;
  }
  __syncthreads();

  const int lane = tid & 31;
  const int wave = tid >> 5;
  const int t0   = (blockIdx.x * 8 + wave) * 16;
  const int col  = lane & 15;
  const int akb  = (lane < 16) ? 0 : 8;
  const int bkb  = (lane < 16) ? 0 : 16;
  const f16* arow = xws + (long)(t0 + col) * XSTRIDE;

  v8f acc[4];
  #pragma unroll
  for (int n = 0; n < 4; ++n)
    #pragma unroll
    for (int r = 0; r < 8; ++r) acc[n][r] = 0.0f;

  #pragma unroll
  for (int kc = 0; kc < 4; ++kc) {
    const int k0 = kc * 32;
    v16h a;
    #pragma unroll
    for (int e = 0; e < 16; ++e) {
      const int K = e + bkb;
      a[e] = arow[k0 + K];
    }
    #pragma unroll
    for (int n = 0; n < 4; ++n) {
      v16h b;
      const f16* bp = &sW[(n * 16 + col) * 130 + k0 + bkb];
      #pragma unroll
      for (int e = 0; e < 16; ++e) b[e] = bp[e];
      acc[n] = __builtin_amdgcn_wmma_f32_16x16x32_f16(
          false, a, false, b, (short)0, acc[n], false, false);
    }
  }

  const int rbase = (lane < 16) ? 0 : 8;
  #pragma unroll
  for (int n = 0; n < 4; ++n) {
    const int g = n * 16 + col;
    const float bias = b_bih[g] + b_bhh[g];
    #pragma unroll
    for (int r = 0; r < 8; ++r)
      xg[(long)(t0 + rbase + r) * G2 + g] = acc[n][r] + bias;
  }
}

// ---------------------------------------------------------------------------
// Kernel 3: beats LSTM scan. thread = sequence (4096), T=8, H=16.
// Writes beats output (first 524288 floats of d_out).
// ---------------------------------------------------------------------------
__global__ __launch_bounds__(256) void k_beats_scan(
    const float* __restrict__ xg, const float* __restrict__ b_whh,
    float* __restrict__ beats_out)
{
  __shared__ float sWhh[64 * 17];
  const int tid = threadIdx.x;
  for (int i = tid; i < 64 * 16; i += 256) {
    const int g = i >> 4, j = i & 15;
    sWhh[g * 17 + j] = b_whh[g * 16 + j];
  }
  __syncthreads();

  const int seq = blockIdx.x * 256 + tid;   // 0..4095
  float h[16], c[16];
  #pragma unroll
  for (int j = 0; j < 16; ++j) { h[j] = 0.0f; c[j] = 0.0f; }

  for (int s = 0; s < 8; ++s) {
    const long tok = (long)seq * 8 + s;
    const float* g = xg + tok * G2;
    float hn[16];
    #pragma unroll 4
    for (int hh = 0; hh < 16; ++hh) {
      float gi = g[hh], gf = g[16 + hh], gg = g[32 + hh], go = g[48 + hh];
      #pragma unroll
      for (int j = 0; j < 16; ++j) {
        const float hj = h[j];
        gi += hj * sWhh[hh * 17 + j];
        gf += hj * sWhh[(16 + hh) * 17 + j];
        gg += hj * sWhh[(32 + hh) * 17 + j];
        go += hj * sWhh[(48 + hh) * 17 + j];
      }
      const float cn = c[hh] * sigf(gf) + sigf(gi) * tanhf(gg);
      c[hh]  = cn;
      hn[hh] = sigf(go) * tanhf(cn);
    }
    float* op = beats_out + tok * H2;
    #pragma unroll
    for (int hh = 0; hh < 16; ++hh) { h[hh] = hn[hh]; op[hh] = hn[hh]; }
  }
}

// ---------------------------------------------------------------------------
// Kernel 4: bars bidirectional LSTM. thread = (seq, dir); dir uniform per block.
// Reads beats[...,7,:] from d_out, writes bars (last 262144 floats of d_out).
// ---------------------------------------------------------------------------
__global__ __launch_bounds__(256) void k_bars_bilstm(
    const float* __restrict__ beats_out,
    const float* __restrict__ f_wih, const float* __restrict__ f_whh,
    const float* __restrict__ f_bih, const float* __restrict__ f_bhh,
    const float* __restrict__ r_wih, const float* __restrict__ r_whh,
    const float* __restrict__ r_bih, const float* __restrict__ r_bhh,
    float* __restrict__ bars_out)
{
  __shared__ float sWih[G3 * 16];
  __shared__ float sWhh[G3 * 33];
  __shared__ float sBias[G3];
  const int tid = threadIdx.x;
  const int gt  = blockIdx.x * 256 + tid;   // 0..2047
  const int dir = blockIdx.x >> 2;          // blocks 0..3 fwd, 4..7 rev (uniform)
  const float* wih = dir ? r_wih : f_wih;
  const float* whh = dir ? r_whh : f_whh;
  const float* bih = dir ? r_bih : f_bih;
  const float* bhh = dir ? r_bhh : f_bhh;
  for (int i = tid; i < G3 * 16; i += 256) sWih[i] = wih[i];
  for (int i = tid; i < G3 * 32; i += 256) {
    const int g = i >> 5, j = i & 31;
    sWhh[g * 33 + j] = whh[i];
  }
  if (tid < G3) sBias[tid] = bih[tid] + bhh[tid];
  __syncthreads();

  const int seq = gt & 1023;
  float h[32], c[32];
  #pragma unroll
  for (int j = 0; j < 32; ++j) { h[j] = 0.0f; c[j] = 0.0f; }

  for (int s = 0; s < 4; ++s) {
    const int t = dir ? (3 - s) : s;
    // xl = beats[b, bar, beat=t, frac=7, :]
    const float* x = beats_out + ((long)(seq * 4 + t) * 8 + 7) * H2;
    float xv[16];
    #pragma unroll
    for (int k = 0; k < 16; ++k) xv[k] = x[k];
    float hn[32];
    for (int hh = 0; hh < 32; ++hh) {
      float gi = sBias[hh], gf = sBias[32 + hh], gg = sBias[64 + hh], go = sBias[96 + hh];
      #pragma unroll
      for (int k = 0; k < 16; ++k) {
        const float xk = xv[k];
        gi += xk * sWih[hh * 16 + k];
        gf += xk * sWih[(32 + hh) * 16 + k];
        gg += xk * sWih[(64 + hh) * 16 + k];
        go += xk * sWih[(96 + hh) * 16 + k];
      }
      #pragma unroll
      for (int j = 0; j < 32; ++j) {
        const float hj = h[j];
        gi += hj * sWhh[hh * 33 + j];
        gf += hj * sWhh[(32 + hh) * 33 + j];
        gg += hj * sWhh[(64 + hh) * 33 + j];
        go += hj * sWhh[(96 + hh) * 33 + j];
      }
      const float cn = c[hh] * sigf(gf) + sigf(gi) * tanhf(gg);
      c[hh]  = cn;
      hn[hh] = sigf(go) * tanhf(cn);
    }
    float* op = bars_out + (long)(seq * 4 + t) * 64 + dir * 32;
    #pragma unroll
    for (int hh = 0; hh < 32; ++hh) { h[hh] = hn[hh]; op[hh] = hn[hh]; }
  }
}

extern "C" void kernel_launch(void* const* d_in, const int* in_sizes, int n_in,
                              void* d_out, int out_size, void* d_ws, size_t ws_size,
                              hipStream_t stream) {
  const float* chan  = (const float*)d_in[0];
  const float* lin_w = (const float*)d_in[1];
  const float* lin_b = (const float*)d_in[2];
  const float* b_wih = (const float*)d_in[3];
  const float* b_whh = (const float*)d_in[4];
  const float* b_bih = (const float*)d_in[5];
  const float* b_bhh = (const float*)d_in[6];
  const float* f_wih = (const float*)d_in[7];
  const float* f_whh = (const float*)d_in[8];
  const float* f_bih = (const float*)d_in[9];
  const float* f_bhh = (const float*)d_in[10];
  const float* r_wih = (const float*)d_in[11];
  const float* r_whh = (const float*)d_in[12];
  const float* r_bih = (const float*)d_in[13];
  const float* r_bhh = (const float*)d_in[14];

  float* out       = (float*)d_out;
  float* beats_out = out;                    // 32768*16 floats
  float* bars_out  = out + BEATS_ELEMS;      // 1024*4*64 floats

  f16*   xws = (f16*)d_ws;                                       // 8 MB
  float* xg  = (float*)((char*)d_ws + (size_t)NTOK * XSTRIDE * 2); // 8 MB

  k_linear_relu <<<256, 256, 0, stream>>>(chan, lin_w, lin_b, xws);
  k_beats_xg    <<<256, 256, 0, stream>>>(xws, b_wih, b_bih, b_bhh, xg);
  k_beats_scan  <<<16,  256, 0, stream>>>(xg, b_whh, beats_out);
  k_bars_bilstm <<<8,   256, 0, stream>>>(beats_out,
                                          f_wih, f_whh, f_bih, f_bhh,
                                          r_wih, r_whh, r_bih, r_bhh,
                                          bars_out);
}
